// RVBProj_normal_8856222565069
// MI455X (gfx1250) — 8-bit matrix operand order repaired and checked by reading, compiled, not run on hardware
//
#include <hip/hip_runtime.h>
#include <hip/hip_bf16.h>

typedef __attribute__((ext_vector_type(8))) int v8i;

#define NSTATE 65536

__constant__ int c_rvb_idx[32] = {
    4005, 4090, 5657, 5702, 10043, 10084, 16007, 16088,
    18713, 18758, 20645, 20730, 24967, 25048, 30779, 30820,
    36241, 36302, 37933, 38002, 42255, 42320, 48307, 48364,
    52013, 52082, 53905, 53966, 58291, 58348, 64015, 64080};

__device__ __forceinline__ double2 cadd(double2 a, double2 b) {
  return make_double2(a.x + b.x, a.y + b.y);
}
__device__ __forceinline__ double2 cmul(double2 a, double2 b) {
  return make_double2(a.x * b.x - a.y * b.y, a.x * b.y + a.y * b.x);
}

__global__ void rvb_init_kernel(double2* __restrict__ st) {
  int j = blockIdx.x * blockDim.x + threadIdx.x;
  double v = 0.0;
#pragma unroll
  for (int k = 0; k < 32; ++k)
    if (j == c_rvb_idx[k]) v = 0.17677669529663687;
  st[j] = make_double2(v, 0.0);
}

__global__ void rvb_step_kernel(const double2* __restrict__ oldv,
                                double2* __restrict__ newv, int shift,
                                const double* __restrict__ z1p,
                                const double* __restrict__ z2p) {
  int j = blockIdx.x * blockDim.x + threadIdx.x;
  int stride = 1 << shift;
  int rp = (j >> shift) & 3;
  int base = j - rp * stride;
  double2 o0 = oldv[base];
  double2 o1 = oldv[base + stride];
  double2 o2 = oldv[base + 2 * stride];
  double2 o3 = oldv[base + 3 * stride];
  double2 S = cadd(cadd(o1, o2), o3);
  double2 z1 = make_double2(z1p[0], z1p[1]);
  double2 z2 = make_double2(z2p[0], z2p[1]);
  double2 r;
  if (rp == 0) {
    r = cadd(o0, cmul(z1, S));
  } else {
    double2 z12 = cmul(z1, z2);
    double2 od = (rp == 1) ? o1 : ((rp == 2) ? o2 : o3);
    r = cadd(cadd(cmul(z2, o0), cmul(z12, S)), od);
  }
  newv[j] = r;
}

__global__ void rvb_clog_kernel(const double2* __restrict__ st,
                                double2* __restrict__ lg) {
  int j = blockIdx.x * blockDim.x + threadIdx.x;
  double2 s = st[j];
  lg[j] = make_double2(0.5 * log(s.x * s.x + s.y * s.y), atan2(s.y, s.x));
}


__device__ __forceinline__ unsigned pack4(int4 s) {
  unsigned u0 = (s.x > 0), u1 = (s.y > 0), u2 = (s.z > 0), u3 = (s.w > 0);
  return u0 | (u1 << 8) | (u2 << 16) | (u3 << 24);
}

__device__ __forceinline__ v8i load_a(const int* __restrict__ x, long row,
                                      bool hihalf) {
  const int4* p = (const int4*)(x + row * 24);
  v8i a = {};
  if (!hihalf) {
    a[0] = (int)pack4(p[0]);
    a[1] = (int)pack4(p[1]);
    a[2] = (int)pack4(p[2]);
    a[3] = (int)pack4(p[3]);
  } else {
    a[0] = (int)pack4(p[4]);
    a[1] = (int)pack4(p[5]);
  }
  return a;
}

__device__ __forceinline__ v8i build_b(int lane) {
  int col = lane & 15;
  int kbase = (lane >> 4) ? 16 : 0;
  v8i b = {};
  if (col < 2) {
#pragma unroll
    for (int v = 0; v < 8; ++v) {
      int kb = kbase + (v & 3) * 4 + ((v >> 2) ? 32 : 0);
      unsigned word = 0;
#pragma unroll
      for (int byte = 0; byte < 4; ++byte) {
        int K = kb + byte;
        unsigned w = 0;
        if (K < 24) {
          int t = K / 3;
          unsigned wk = (unsigned)(3 - (K % 3)) << (2 * (7 - t));
          w = (col == 0) ? (wk & 0xFFu) : ((wk >> 8) & 0xFFu);
        }
        word |= w << (8 * byte);
      }
      b[v] = (int)word;
    }
  }
  return b;
}

__global__ __launch_bounds__(256) void rvb_gather_kernel(
    const int* __restrict__ x, const double2* __restrict__ logtab,
    double2* __restrict__ out, long rows) {
  __shared__ int lds_lo[8][32];
  __shared__ int lds_hi[8][32];

  const int lane = threadIdx.x & 31;
  const int wave = threadIdx.x >> 5;
  const long tileBase = (long)blockIdx.x * 256 + (long)wave * 32;
  const bool hihalf = (lane >= 16);
  const long rmax = rows - 1;

  v8i bmat = build_b(lane);

  long r0 = tileBase + (lane & 15);
  long r1 = r0 + 16;
  if (r0 > rmax) r0 = rmax;
  if (r1 > rmax) r1 = rmax;
  v8i a0 = load_a(x, r0, hihalf);
  v8i a1 = load_a(x, r1, hihalf);

  v8i czero = {};
  v8i d0 = __builtin_amdgcn_wmma_i32_16x16x64_iu8(false, a0, false, bmat,
                                                  czero, false, false);
  v8i d1 = __builtin_amdgcn_wmma_i32_16x16x64_iu8(false, a1, false, bmat,
                                                  czero, false, false);

  const int col = lane & 15;
  if (col < 2) {
    const int half = (lane >> 4) & 1;
    int* dst = (col == 0) ? lds_lo[wave] : lds_hi[wave];
#pragma unroll
    for (int m = 0; m < 8; ++m) dst[half * 8 + m] = d0[m];
#pragma unroll
    for (int m = 0; m < 8; ++m) dst[16 + half * 8 + m] = d1[m];
  }

  const int idx = lds_lo[wave][lane] + (lds_hi[wave][lane] << 8);
  const long row = tileBase + lane;
  double2 lv = logtab[idx & (NSTATE - 1)];
  if (row < rows) out[row] = lv;
}

extern "C" void kernel_launch(void* const* d_in, const int* in_sizes, int n_in,
                              void* d_out, int out_size, void* d_ws,
                              size_t ws_size, hipStream_t stream) {
  const int* x_in = (const int*)d_in[0];
  const double* z1 = (const double*)d_in[1];
  const double* z2 = (const double*)d_in[2];
  double2* out = (double2*)d_out;

  char* ws = (char*)d_ws;
  double2* bufA = (double2*)ws;
  double2* bufB = (double2*)(ws + (size_t)NSTATE * sizeof(double2));
  double2* logtab = (double2*)(ws + 2 * (size_t)NSTATE * sizeof(double2));

  rvb_init_kernel<<<NSTATE / 256, 256, 0, stream>>>(bufA);

  double2* src = bufA;
  double2* dst = bufB;
  for (int t = 0; t < 8; ++t) {
    rvb_step_kernel<<<NSTATE / 256, 256, 0, stream>>>(src, dst, 2 * (7 - t),
                                                      z1, z2);
    double2* tmp = src; src = dst; dst = tmp;
  }
  rvb_clog_kernel<<<NSTATE / 256, 256, 0, stream>>>(src, logtab);

  long rows = (long)in_sizes[0] / 24;
  long blocks = (rows + 255) / 256;
  rvb_gather_kernel<<<(int)blocks, 256, 0, stream>>>(x_in, logtab, out, rows);
}
